// Convpass_swin_hypernet_fusev2_22359599743390
// MI455X (gfx1250) — matrix operand order repaired and checked by reading, not compiled, not run on hardware
//
#include <hip/hip_runtime.h>
#include <hip/hip_bf16.h>

typedef _Float16 half_t;
typedef __attribute__((ext_vector_type(16))) _Float16 v16h;
typedef __attribute__((ext_vector_type(8)))  _Float16 v8h;
typedef __attribute__((ext_vector_type(8)))  float    v8f;
typedef __attribute__((ext_vector_type(4)))  unsigned int u32x4;
typedef __attribute__((ext_vector_type(8)))  int i32x8;
typedef __attribute__((ext_vector_type(4)))  int i32x4;

#define NB 32
#define HH 56
#define WW 56
#define CC 768
#define DIMC 128
#define NTOK (NB*HH*WW)          // 100352
#define MTILES (NTOK/16)         // 6272
#define HW (HH*WW)               // 3136

__device__ __forceinline__ v8f wmma16(v16h a, v16h b, v8f c) {
  return __builtin_amdgcn_wmma_f32_16x16x32_f16(
      false, a, false, b, (short)0, c, false, false);
}

// A fragment (16x32 f16) from an f32 row-major row, converting on the fly.
__device__ __forceinline__ v16h load_a_f32(const float* __restrict__ row,
                                           int kbase, int khalf) {
  const float4 a0 = *(const float4*)(row + kbase + 8*khalf);
  const float4 a1 = *(const float4*)(row + kbase + 8*khalf + 4);
  const float4 b0 = *(const float4*)(row + kbase + 16 + 8*khalf);
  const float4 b1 = *(const float4*)(row + kbase + 16 + 8*khalf + 4);
  v16h r;
  r[0]=(half_t)a0.x;  r[1]=(half_t)a0.y;  r[2]=(half_t)a0.z;  r[3]=(half_t)a0.w;
  r[4]=(half_t)a1.x;  r[5]=(half_t)a1.y;  r[6]=(half_t)a1.z;  r[7]=(half_t)a1.w;
  r[8]=(half_t)b0.x;  r[9]=(half_t)b0.y;  r[10]=(half_t)b0.z; r[11]=(half_t)b0.w;
  r[12]=(half_t)b1.x; r[13]=(half_t)b1.y; r[14]=(half_t)b1.z; r[15]=(half_t)b1.w;
  return r;
}

__device__ __forceinline__ v16h load_a_f16(const half_t* __restrict__ row,
                                           int kbase, int khalf) {
  v8h lo = *(const v8h*)(row + kbase + 8*khalf);
  v8h hi = *(const v8h*)(row + kbase + 16 + 8*khalf);
  return __builtin_shufflevector(lo, hi, 0,1,2,3,4,5,6,7,8,9,10,11,12,13,14,15);
}

__device__ __forceinline__ float quick_gelu(float v) {
  return v / (1.0f + __expf(-1.702f * v));
}

// ---- Tensor Data Mover: 2D tile (f16 elements) global -> LDS -------------
// tile_d0 elements per row, tile_d1 rows, row stride stride_d0 elements.
// D# per CDNA5 ISA 8.3/8.4: count=1, type=2, data_size=1 (2 bytes).
__device__ __forceinline__ void tdm_load_2d_f16(unsigned int lds_off,
                                                const void* gsrc,
                                                unsigned int tile_d0,
                                                unsigned int tile_d1,
                                                unsigned int stride_d0) {
  unsigned long long ga = (unsigned long long)gsrc;
  unsigned int td0 = stride_d0;          // tensor_dim0 >= tile width
  unsigned int td1 = tile_d1;            // tensor_dim1
  u32x4 g0;
  g0[0] = 1u;                                            // count=1, user D#
  g0[1] = lds_off;                                       // lds_addr (bytes)
  g0[2] = (unsigned int)(ga & 0xffffffffu);              // global_addr[31:0]
  g0[3] = (unsigned int)((ga >> 32) & 0x01ffffffu) | (2u << 30); // [56:32]|type=2
  i32x8 g1;
  g1[0] = (int)(1u << 16);                               // data_size=1 (2B)
  g1[1] = (int)((td0 & 0xffffu) << 16);                  // tensor_dim0[15:0]
  g1[2] = (int)((td0 >> 16) | ((td1 & 0xffffu) << 16));  // td0 hi | td1 lo
  g1[3] = (int)((td1 >> 16) | ((tile_d0 & 0xffffu) << 16)); // td1 hi | tile_dim0
  g1[4] = (int)(tile_d1 & 0xffffu);                      // tile_dim1, tile_dim2=0
  g1[5] = (int)stride_d0;                                // tensor_dim0_stride lo
  g1[6] = 0;                                             // stride hi | dim1_stride lo
  g1[7] = 0;
  i32x4 gz = {0, 0, 0, 0};
#if __clang_major__ >= 23
  i32x8 gz8 = {0, 0, 0, 0, 0, 0, 0, 0};
  __builtin_amdgcn_tensor_load_to_lds(g0, g1, gz, gz, gz8, 0);
#else
  __builtin_amdgcn_tensor_load_to_lds(g0, g1, gz, gz, 0);
#endif
}

// ---------------- init ----------------
__global__ void zero64_kernel(float* p) { p[threadIdx.x] = 0.0f; }

// ---------------- weight pack: f32 [K,N] row-major -> WMMA B fragments ----
__global__ void pack_b_kernel(const float* __restrict__ src,
                              half_t* __restrict__ dst, int K, int N) {
  int idx = blockIdx.x * blockDim.x + threadIdx.x;
  if (idx >= K * N) return;
  int k = idx / N, n = idx % N;
  int kstep = k >> 5, i32 = k & 31;
  int khalf = (i32 >> 3) & 1, e = (i32 & 7) | ((i32 >> 4) << 3);
  int ntile = n >> 4;
  int lane = khalf * 16 + (n & 15);
  size_t off = ((size_t)(kstep * (N >> 4) + ntile)) * 512 + lane * 16 + e;
  dst[off] = (half_t)src[idx];
}

// ------------- fused: down-proj + QuickGELU  AND  meta relu-sum ----------
__global__ void down_meta_kernel(const float* __restrict__ x,
                                 const half_t* __restrict__ dwp,
                                 const half_t* __restrict__ w1p,
                                 const float* __restrict__ db,
                                 const float* __restrict__ b1,
                                 half_t* __restrict__ xdown,
                                 float* __restrict__ sum64) {
  int wave = threadIdx.x >> 5, lane = threadIdx.x & 31;
  int n = lane & 15, khalf = lane >> 4;
  int mtile = blockIdx.x * 8 + wave;
  const float* row = x + ((size_t)mtile * 16 + n) * CC;   // m == lane&15

  v8f z = {};
  v8f d[8], mz[4];
#pragma unroll
  for (int q = 0; q < 8; ++q) d[q] = z;
#pragma unroll
  for (int q = 0; q < 4; ++q) mz[q] = z;

  for (int ks = 0; ks < 24; ++ks) {
    v16h a = load_a_f32(row, ks * 32, khalf);
    v16h bb[6];
    // batch 1: down ntiles 0..5 (loads issued together, one wait)
#pragma unroll
    for (int q = 0; q < 6; ++q)
      bb[q] = *(const v16h*)(dwp + ((size_t)(ks * 8 + q)) * 512 + lane * 16);
#pragma unroll
    for (int q = 0; q < 6; ++q) d[q] = wmma16(a, bb[q], d[q]);
    // batch 2: down ntiles 6..7 + meta ntiles 0..3
#pragma unroll
    for (int q = 0; q < 2; ++q)
      bb[q] = *(const v16h*)(dwp + ((size_t)(ks * 8 + 6 + q)) * 512 + lane * 16);
#pragma unroll
    for (int q = 0; q < 4; ++q)
      bb[2 + q] = *(const v16h*)(w1p + ((size_t)(ks * 4 + q)) * 512 + lane * 16);
#pragma unroll
    for (int q = 0; q < 2; ++q) d[6 + q] = wmma16(a, bb[q], d[6 + q]);
#pragma unroll
    for (int q = 0; q < 4; ++q) mz[q] = wmma16(a, bb[2 + q], mz[q]);
  }

  // down epilogue: bias + QuickGELU -> f16 [NTOK,128]
#pragma unroll
  for (int q = 0; q < 8; ++q) {
    int col = q * 16 + n;
    float bias = db[col];
#pragma unroll
    for (int r = 0; r < 8; ++r) {
      int m = mtile * 16 + r + 8 * khalf;
      xdown[(size_t)m * DIMC + col] = (half_t)quick_gelu(d[q][r] + bias);
    }
  }
  // meta epilogue
#pragma unroll
  for (int q = 0; q < 4; ++q) {
    float bias = b1[q * 16 + n];
    float part = 0.0f;
#pragma unroll
    for (int r = 0; r < 8; ++r) part += fmaxf(mz[q][r] + bias, 0.0f);
    part += __shfl_xor(part, 16, 32);
    if (khalf == 0) atomicAdd(&sum64[q * 16 + n], part);
  }
}

// ---------------- hypernet: prompt -> conv weight, packed for WMMA B ------
__global__ void hypernet_kernel(const float* __restrict__ sum64,
                                const float* __restrict__ w2,
                                const float* __restrict__ b2,
                                const float* __restrict__ layer_emb,
                                const float* __restrict__ hyper_w,
                                const float* __restrict__ hyper_b,
                                half_t* __restrict__ cwp) {
  __shared__ float tvec[64];
  int tid = threadIdx.x;
  if (tid < 64) {
    float acc = b2[tid];
    const float inv_n = 1.0f / (float)NTOK;
    for (int k = 0; k < 64; ++k)
      acc += (sum64[k] * inv_n) * w2[k * 64 + tid];
    tvec[tid] = layer_emb[tid] + acc;
  }
  __syncthreads();
  int j = blockIdx.x * blockDim.x + tid;       // exact grid over O*I*3*3
  float acc = hyper_b[j];
  for (int k = 0; k < 64; ++k)
    acc += tvec[k] * hyper_w[(size_t)k * (DIMC * DIMC * 9) + j];
  int kw = j % 3, t1 = j / 3, kh = t1 % 3, t2 = t1 / 3;
  int i = t2 & 127, o = t2 >> 7;
  int tap = kh * 3 + kw;
  int kstep = i >> 5, i32 = i & 31;
  int khalf = (i32 >> 3) & 1, e = (i32 & 7) | ((i32 >> 4) << 3);
  int ntile = o >> 4;
  int lane = khalf * 16 + (o & 15);
  size_t off = (((size_t)tap * 4 + kstep) * 8 + ntile) * 512 + lane * 16 + e;
  cwp[off] = (half_t)acc;
}

// ---------------- 3x3 conv (implicit GEMM) + QuickGELU -> f16 [NTOK,128] --
// Per-tap B panel (16 KB = 4 rows x 4 KB, row stride 8 KB) staged via TDM.
__global__ void conv_gelu_kernel(const half_t* __restrict__ xdown,
                                 const half_t* __restrict__ cwp,
                                 half_t* __restrict__ y) {
  __shared__ half_t bsm[16 * 512];               // 4 kk * 4 ntiles * 512 f16
  int wave = threadIdx.x >> 5, lane = threadIdx.x & 31;
  int n = lane & 15, khalf = lane >> 4;
  int mtile = blockIdx.x * 8 + wave;
  int nbase = blockIdx.y * 4;
  int t = mtile * 16 + n;
  int bimg = t / HW;
  int rem = t - bimg * HW;
  int hh = rem / WW;
  int ww = rem - hh * WW;
  v8f z = {};
  v16h zh = {};
  v8f acc[4];
#pragma unroll
  for (int q = 0; q < 4; ++q) acc[q] = z;

  for (int tap = 0; tap < 9; ++tap) {
    __syncthreads();                             // previous tap's LDS reads done
    if (threadIdx.x < 32) {                      // wave 0 issues the tensor DMA
      tdm_load_2d_f16((unsigned int)(size_t)&bsm[0],
                      cwp + (((size_t)tap * 4) * 8 + nbase) * 512,
                      /*tile_d0=*/2048, /*tile_d1=*/4, /*stride_d0=*/4096);
      __builtin_amdgcn_s_wait_tensorcnt(0);
    }
    __syncthreads();                             // B panel visible to all waves

    int h2 = hh + tap / 3 - 1;
    int w2 = ww + tap % 3 - 1;
    bool valid = ((unsigned)h2 < (unsigned)HH) && ((unsigned)w2 < (unsigned)WW);
    const half_t* row = xdown + ((size_t)((bimg * HH + h2) * WW + w2)) * DIMC;
    if (tap < 8) {
      int h3 = hh + (tap + 1) / 3 - 1;
      int w3 = ww + (tap + 1) % 3 - 1;
      if (((unsigned)h3 < (unsigned)HH) && ((unsigned)w3 < (unsigned)WW))
        __builtin_prefetch(xdown + ((size_t)((bimg * HH + h3) * WW + w3)) * DIMC, 0, 1);
    }
    v16h af[4];
#pragma unroll
    for (int kk = 0; kk < 4; ++kk) {
      af[kk] = zh;
      if (valid) af[kk] = load_a_f16(row, kk * 32, khalf);
    }
#pragma unroll
    for (int kk = 0; kk < 4; ++kk) {
#pragma unroll
      for (int q = 0; q < 4; ++q) {
        v16h b = *(const v16h*)(&bsm[(kk * 4 + q) * 512 + lane * 16]);
        acc[q] = wmma16(af[kk], b, acc[q]);
      }
    }
  }
#pragma unroll
  for (int q = 0; q < 4; ++q) {
    int col = (nbase + q) * 16 + n;
#pragma unroll
    for (int r = 0; r < 8; ++r) {
      int m = mtile * 16 + r + 8 * khalf;
      y[(size_t)m * DIMC + col] = (half_t)quick_gelu(acc[q][r]);
    }
  }
}

// ---------------- up proj + bias -> f32 output [NTOK,768] -----------------
// B panel (16 KB = 4 rows x 4 KB, row stride 48 KB) staged once via TDM.
__global__ void up_proj_kernel(const half_t* __restrict__ y,
                               const half_t* __restrict__ uwp,
                               const float* __restrict__ ub,
                               float* __restrict__ out) {
  __shared__ half_t bsm[16 * 512];               // 4 ksteps * 4 ntiles
  int wave = threadIdx.x >> 5, lane = threadIdx.x & 31;
  int n = lane & 15, khalf = lane >> 4;
  int mtile = blockIdx.x * 8 + wave;
  int nbase = blockIdx.y * 4;                    // N=768 -> 48 ntiles, grid.y=12
  if (threadIdx.x < 32) {
    tdm_load_2d_f16((unsigned int)(size_t)&bsm[0],
                    uwp + ((size_t)nbase) * 512,
                    /*tile_d0=*/2048, /*tile_d1=*/4, /*stride_d0=*/24576);
    __builtin_amdgcn_s_wait_tensorcnt(0);
  }
  __syncthreads();

  const half_t* row = y + ((size_t)mtile * 16 + n) * DIMC;
  v16h af[4];
#pragma unroll
  for (int ks = 0; ks < 4; ++ks) af[ks] = load_a_f16(row, ks * 32, khalf);

  v8f c0 = {}, c1 = {}, c2 = {}, c3 = {};
#pragma unroll
  for (int ks = 0; ks < 4; ++ks) {               // K = 128
    v16h b0 = *(const v16h*)(&bsm[(ks * 4 + 0) * 512 + lane * 16]);
    v16h b1 = *(const v16h*)(&bsm[(ks * 4 + 1) * 512 + lane * 16]);
    v16h b2 = *(const v16h*)(&bsm[(ks * 4 + 2) * 512 + lane * 16]);
    v16h b3 = *(const v16h*)(&bsm[(ks * 4 + 3) * 512 + lane * 16]);
    c0 = wmma16(af[ks], b0, c0);
    c1 = wmma16(af[ks], b1, c1);
    c2 = wmma16(af[ks], b2, c2);
    c3 = wmma16(af[ks], b3, c3);
  }
  v8f acc[4] = {c0, c1, c2, c3};
#pragma unroll
  for (int q = 0; q < 4; ++q) {
    int col = (nbase + q) * 16 + n;
    float bias = ub[col];
#pragma unroll
    for (int r = 0; r < 8; ++r) {
      int m = mtile * 16 + r + 8 * khalf;
      out[(size_t)m * CC + col] = acc[q][r] + bias;
    }
  }
}

extern "C" void kernel_launch(void* const* d_in, const int* in_sizes, int n_in,
                              void* d_out, int out_size, void* d_ws, size_t ws_size,
                              hipStream_t stream) {
  const float* x         = (const float*)d_in[0];
  const float* meta_w1   = (const float*)d_in[1];
  const float* meta_b1   = (const float*)d_in[2];
  const float* meta_w2   = (const float*)d_in[3];
  const float* meta_b2   = (const float*)d_in[4];
  const float* layer_emb = (const float*)d_in[5];
  const float* hyper_w   = (const float*)d_in[6];
  const float* hyper_b   = (const float*)d_in[7];
  const float* down_w    = (const float*)d_in[8];
  const float* down_b    = (const float*)d_in[9];
  const float* up_w      = (const float*)d_in[10];
  const float* up_b      = (const float*)d_in[11];
  float* out = (float*)d_out;

  char* ws = (char*)d_ws;
  size_t off = 0;
  auto alloc = [&](size_t bytes) { char* p = ws + off; off = (off + bytes + 255) & ~(size_t)255; return p; };
  float*  sum64 = (float*) alloc(64 * sizeof(float));
  half_t* w1p   = (half_t*)alloc((size_t)CC * 64 * sizeof(half_t));
  half_t* dwp   = (half_t*)alloc((size_t)CC * DIMC * sizeof(half_t));
  half_t* uwp   = (half_t*)alloc((size_t)DIMC * CC * sizeof(half_t));
  half_t* cwp   = (half_t*)alloc((size_t)DIMC * DIMC * 9 * sizeof(half_t));
  half_t* xdown = (half_t*)alloc((size_t)NTOK * DIMC * sizeof(half_t));
  half_t* yact  = (half_t*)alloc((size_t)NTOK * DIMC * sizeof(half_t));
  (void)ws_size; (void)in_sizes; (void)n_in; (void)out_size;

  zero64_kernel<<<1, 64, 0, stream>>>(sum64);

  pack_b_kernel<<<(CC * 64 + 255) / 256, 256, 0, stream>>>(meta_w1, w1p, CC, 64);
  pack_b_kernel<<<(CC * DIMC + 255) / 256, 256, 0, stream>>>(down_w, dwp, CC, DIMC);
  pack_b_kernel<<<(DIMC * CC + 255) / 256, 256, 0, stream>>>(up_w, uwp, DIMC, CC);

  down_meta_kernel<<<MTILES / 8, 256, 0, stream>>>(x, dwp, w1p, down_b, meta_b1,
                                                   xdown, sum64);

  hypernet_kernel<<<(DIMC * DIMC * 9) / 256, 256, 0, stream>>>(
      sum64, meta_w2, meta_b2, layer_emb, hyper_w, hyper_b, cwp);

  conv_gelu_kernel<<<dim3(MTILES / 8, 2), 256, 0, stream>>>(xdown, cwp, yact);

  up_proj_kernel<<<dim3(MTILES / 8, 12), 256, 0, stream>>>(yact, uwp, up_b, out);
}
